// LlamaAttention_70918499991961
// MI455X (gfx1250) — matrix operand order repaired and checked by reading, not compiled, not run on hardware
//
#include <hip/hip_runtime.h>

// ---------- types ----------
typedef __attribute__((ext_vector_type(4)))  unsigned int u32x4;
typedef __attribute__((ext_vector_type(4)))  int          i32x4;
typedef __attribute__((ext_vector_type(8)))  int          i32x8;
typedef __attribute__((ext_vector_type(8)))  __bf16       bf16x8;
typedef __attribute__((ext_vector_type(16))) __bf16       bf16x16;
typedef __attribute__((ext_vector_type(8)))  float        f32x8;

// ---------- helpers ----------
static __device__ __forceinline__ unsigned short f32_to_bf16(float f) {
    unsigned int u = __float_as_uint(f);
    u += 0x7FFFu + ((u >> 16) & 1u);   // round-to-nearest-even
    return (unsigned short)(u >> 16);
}
static __device__ __forceinline__ float bf16_to_f32(unsigned short u) {
    return __uint_as_float(((unsigned int)u) << 16);
}
static __device__ __forceinline__ f32x8 zero8() {
    f32x8 z;
#pragma unroll
    for (int i = 0; i < 8; ++i) z[i] = 0.0f;
    return z;
}
// Build a v16 bf16 fragment from two aligned 16-byte chunks.
static __device__ __forceinline__ bf16x16 make_frag(const unsigned short* p0,
                                                    const unsigned short* p1) {
    u32x4 lo = *(const u32x4*)p0;
    u32x4 hi = *(const u32x4*)p1;
    bf16x8 a = __builtin_bit_cast(bf16x8, lo);
    bf16x8 b = __builtin_bit_cast(bf16x8, hi);
    return __builtin_shufflevector(a, b, 0,1,2,3,4,5,6,7,8,9,10,11,12,13,14,15);
}
static __device__ __forceinline__ f32x8 wmma_bf16(bf16x16 a, bf16x16 b, f32x8 c) {
    return __builtin_amdgcn_wmma_f32_16x16x32_bf16(false, a, false, b,
                                                   (short)0, c, false, false);
}

// DPP16 butterfly reductions across a 16-lane group (replaces ds_bpermute).
// xor1 -> quad_perm[1,0,3,2]=0xB1; xor2 -> quad_perm[2,3,0,1]=0x4E;
// after those each quad is uniform so ^7 (row_half_mirror=0x141) == ^4 and
// ^15 (row_mirror=0x140) == ^8.
template <int CTRL>
static __device__ __forceinline__ float dpp_f32(float x) {
    int r = __builtin_amdgcn_update_dpp(0, __float_as_int(x), CTRL, 0xF, 0xF, true);
    return __int_as_float(r);
}
static __device__ __forceinline__ float red_max16(float x) {
    x = fmaxf(x, dpp_f32<0xB1>(x));
    x = fmaxf(x, dpp_f32<0x4E>(x));
    x = fmaxf(x, dpp_f32<0x141>(x));
    x = fmaxf(x, dpp_f32<0x140>(x));
    return x;
}
static __device__ __forceinline__ float red_sum16(float x) {
    x += dpp_f32<0xB1>(x);
    x += dpp_f32<0x4E>(x);
    x += dpp_f32<0x141>(x);
    x += dpp_f32<0x140>(x);
    return x;
}

// ---------- fp32 -> bf16 convert ----------
__global__ void f32_to_bf16_kernel(const float* __restrict__ in,
                                   unsigned short* __restrict__ out, size_t n) {
    size_t i = (size_t)blockIdx.x * blockDim.x + threadIdx.x;
    size_t stride = (size_t)gridDim.x * blockDim.x;
    for (; i < n; i += stride) out[i] = f32_to_bf16(in[i]);
}

// ---------- RoPE (in-place on bf16 [B,S,H,128]) ----------
__global__ void rope_kernel(unsigned short* __restrict__ x,
                            const long long* __restrict__ pos_ids,
                            int S, int H, size_t total) {
    size_t idx = (size_t)blockIdx.x * blockDim.x + threadIdx.x;
    if (idx >= total) return;
    int hd = (int)(idx & 63);             // rotary pair index 0..63
    size_t rem = idx >> 6;
    int h = (int)(rem % H);
    size_t bs = rem / H;                  // 0 .. B*S-1
    size_t s = bs % S, b = bs / S;
    float pos = (float)pos_ids[b * S + s];
    // inv_freq = 10000^(-hd/64) = exp(-hd * ln(1e4)/64)
    float ang = pos * __expf(-(float)hd * 0.14391156f);
    float sn, c;
    __sincosf(ang, &sn, &c);
    unsigned short* p = x + (bs * H + h) * 128;
    float x1 = bf16_to_f32(p[hd]);
    float x2 = bf16_to_f32(p[hd + 64]);
    p[hd]      = f32_to_bf16(x1 * c - x2 * sn);
    p[hd + 64] = f32_to_bf16(x2 * c + x1 * sn);
}

// ---------- bf16 WMMA GEMM:  C[MxN] = A[MxK] * B[KxN] ----------
// 256 threads = 8 waves; block tile 128x128, K-step 32, double-buffered LDS.
// Wave grid 4x2; each wave: 32x64 = 2x4 tiles of 16x16.
template <bool STORE_BF16>
__global__ __launch_bounds__(256)
void gemm_bf16_kernel(const unsigned short* __restrict__ A,
                      const unsigned short* __restrict__ B,
                      void* __restrict__ C, int M, int N, int K) {
    __shared__ __align__(16) unsigned short As[2][128][40];   // [m][k]
    __shared__ __align__(16) unsigned short Bt[2][128][40];   // B^T [n][k]

    const int tid  = threadIdx.x;
    const int lane = tid & 31;
    const int wave = tid >> 5;
    const int wm   = wave >> 1;        // 0..3
    const int wn   = wave & 1;         // 0..1
    const int g    = lane >> 4;        // lane group
    const int ln   = lane & 15;

    const int block_m = blockIdx.y * 128;
    const int block_n = blockIdx.x * 128;

    f32x8 acc[2][4];
#pragma unroll
    for (int i = 0; i < 2; ++i)
#pragma unroll
        for (int j = 0; j < 4; ++j) acc[i][j] = zero8();

    const int arow = tid >> 1, acol = (tid & 1) << 4;   // A: 16 halves/thread
    const int krow = tid >> 3, ncol = (tid & 7) << 4;   // B: 16 halves/thread

    auto stage = [&](int buf, int k0) {
        const unsigned short* asrc = A + (size_t)(block_m + arow) * K + k0 + acol;
        *(u32x4*)&As[buf][arow][acol]     = *(const u32x4*)asrc;
        *(u32x4*)&As[buf][arow][acol + 8] = *(const u32x4*)(asrc + 8);
        const unsigned short* bsrc = B + (size_t)(k0 + krow) * N + block_n + ncol;
        u32x4 d0 = *(const u32x4*)bsrc;
        u32x4 d1 = *(const u32x4*)(bsrc + 8);
#pragma unroll
        for (int w = 0; w < 4; ++w) {
            Bt[buf][ncol + 2 * w    ][krow] = (unsigned short)(d0[w]);
            Bt[buf][ncol + 2 * w + 1][krow] = (unsigned short)(d0[w] >> 16);
            Bt[buf][ncol + 8 + 2 * w][krow] = (unsigned short)(d1[w]);
            Bt[buf][ncol + 9 + 2 * w][krow] = (unsigned short)(d1[w] >> 16);
        }
    };

    stage(0, 0);
    __syncthreads();

    int buf = 0;
    for (int k0 = 0; k0 < K; k0 += 32) {
        const int knext = k0 + 32;
        if (knext < K) {
            __builtin_prefetch(A + (size_t)(block_m + arow) * K + knext + acol, 0, 0);
            stage(buf ^ 1, knext);   // overlaps with WMMAs on current buffer
        }

        bf16x16 afrag[2], bfrag[4];
#pragma unroll
        for (int i = 0; i < 2; ++i) {
            const int row = wm * 32 + i * 16 + ln;
            afrag[i] = make_frag(&As[buf][row][8 * g], &As[buf][row][16 + 8 * g]);
        }
#pragma unroll
        for (int j = 0; j < 4; ++j) {
            const int n = wn * 64 + j * 16 + ln;
            bfrag[j] = make_frag(&Bt[buf][n][8 * g], &Bt[buf][n][16 + 8 * g]);
        }
#pragma unroll
        for (int i = 0; i < 2; ++i)
#pragma unroll
            for (int j = 0; j < 4; ++j)
                acc[i][j] = wmma_bf16(afrag[i], bfrag[j], acc[i][j]);
        __syncthreads();
        buf ^= 1;
    }

    // ---- epilogue ----
#pragma unroll
    for (int i = 0; i < 2; ++i) {
#pragma unroll
        for (int j = 0; j < 4; ++j) {
            const int mbase = block_m + wm * 32 + i * 16 + 8 * g;
            const int col   = block_n + wn * 64 + j * 16 + ln;
#pragma unroll
            for (int r = 0; r < 8; ++r) {
                if (STORE_BF16)
                    ((unsigned short*)C)[(size_t)(mbase + r) * N + col] =
                        f32_to_bf16(acc[i][j][r]);
                else
                    ((float*)C)[(size_t)(mbase + r) * N + col] = acc[i][j][r];
            }
        }
    }
}

// ---------- Flash attention (bf16 in, bf16 ctx out, fp32 online softmax) ----------
// grid: (S/64, 32 heads, B); block: 128 threads = 4 waves; wave = one 16-row q tile.
// K tile is staged LDS-side by the Tensor Data Mover (wave 0 issues the DMA with an
// LDS-padding descriptor matching Ks's 272-byte row pitch); V is transposed by the
// vector path in parallel.
__global__ __launch_bounds__(128)
void attn_kernel(const unsigned short* __restrict__ Q,   // [B,S,32,128]
                 const unsigned short* __restrict__ Kp,  // [B,S,8,128]
                 const unsigned short* __restrict__ Vp,  // [B,S,8,128]
                 unsigned short* __restrict__ Ctx,       // [B,S,32,128]
                 int S) {
    __shared__ __align__(16) unsigned short Ks[32][136];     // [kv][d]  (256B+16B pad rows)
    __shared__ __align__(16) unsigned short Vt[128][40];     // V^T [d][kv]
    __shared__ __align__(16) unsigned short Ps[4][16][40];   // per-wave P [m][k]

    const int tid  = threadIdx.x;
    const int lane = tid & 31;
    const int wave = tid >> 5;
    const int g    = lane >> 4;
    const int ln   = lane & 15;
    const int b    = blockIdx.z;
    const int h    = blockIdx.y;
    const int kh   = h >> 2;                 // GQA: 4 q-heads per kv-head
    const int qblk = blockIdx.x;
    const int q0   = qblk * 64 + wave * 16;
    const float scale = 0.08838834764831845f;   // 1/sqrt(128)

    // ---- TDM descriptor (loop-invariant part): 2D tile, 4-byte elements ----
    // data_size=2(4B), pad_enable, pad_interval=5 (pad every 64 DWORDs=256B),
    // pad_amount=3 (4 DWORDs=16B) -> LDS row pitch 272B == sizeof(Ks[0]).
    i32x8 g1;
    g1[0] = (int)((2u << 16) | (1u << 20) | (5u << 22) | (3u << 25));
    g1[1] = (int)(64u << 16);    // tensor_dim0 = 64 DWORDs (one 256B row)
    g1[2] = (int)(32u << 16);    // tensor_dim1 = 32 rows
    g1[3] = (int)(64u << 16);    // tile_dim0   = 64 DWORDs
    g1[4] = 32;                  // tile_dim1   = 32, tile_dim2 = 0
    g1[5] = 512;                 // tensor_dim0_stride = 2048B = 512 DWORDs (8 kv-heads * 256B)
    g1[6] = 0;
    g1[7] = 0;
    i32x4 gz4;
    gz4[0] = gz4[1] = gz4[2] = gz4[3] = 0;
    i32x8 gz8;
#pragma unroll
    for (int i = 0; i < 8; ++i) gz8[i] = 0;
    const unsigned ldsKs = (unsigned)(uintptr_t)&Ks[0][0];  // flat LDS addr[31:0] = LDS offset

    // Q fragments: 16x128 = 4 x (16x32) bf16
    bf16x16 qfrag[4];
    {
        const unsigned short* qp =
            Q + (((size_t)b * S + (q0 + ln)) * 32 + h) * 128;
#pragma unroll
        for (int st = 0; st < 4; ++st)
            qfrag[st] = make_frag(qp + st * 32 + 8 * g, qp + st * 32 + 16 + 8 * g);
    }

    float m_r[8], l_r[8];
    f32x8 oacc[8];
#pragma unroll
    for (int r = 0; r < 8; ++r) { m_r[r] = -1e30f; l_r[r] = 0.0f; }
#pragma unroll
    for (int dt = 0; dt < 8; ++dt) oacc[dt] = zero8();

    const int nblk = 2 * qblk + 2;                 // kv covers 0 .. q0_block+63
    const int lrow = tid >> 2, lcol = (tid & 3) << 5;
    for (int kb = 0; kb < nblk; ++kb) {
        const int kv0 = kb * 32;

        // ---- issue TDM for the K tile (wave 0 only) ----
        if (wave == 0) {
            unsigned long long ga = (unsigned long long)(uintptr_t)
                (Kp + (((size_t)b * S + kv0) * 8 + kh) * 128);
            u32x4 g0;
            g0[0] = 1u;                               // count=1 (valid user D#)
            g0[1] = ldsKs;                            // lds_addr
            g0[2] = (unsigned)ga;                     // global_addr[31:0]
            g0[3] = ((unsigned)(ga >> 32) & 0x01FFFFFFu) | (2u << 30);  // addr[56:32] | type=2
            __builtin_amdgcn_tensor_load_to_lds(g0, g1, gz4, gz4, gz8, 0);
        }

        // ---- stage V transposed (all threads, overlaps the TDM) ----
        {
            const unsigned short* vsrc =
                Vp + (((size_t)b * S + kv0 + lrow) * 8 + kh) * 128 + lcol;
#pragma unroll
            for (int c4 = 0; c4 < 4; ++c4) {
                u32x4 vd = *(const u32x4*)(vsrc + c4 * 8);
#pragma unroll
                for (int w = 0; w < 4; ++w) {
                    Vt[lcol + c4 * 8 + 2 * w    ][lrow] = (unsigned short)(vd[w]);
                    Vt[lcol + c4 * 8 + 2 * w + 1][lrow] = (unsigned short)(vd[w] >> 16);
                }
            }
        }
        if (wave == 0) __builtin_amdgcn_s_wait_tensorcnt((short)0);
        __syncthreads();

        // ---- two 16x16 score tiles ----
#pragma unroll
        for (int t = 0; t < 2; ++t) {
            f32x8 sacc = zero8();
#pragma unroll
            for (int st = 0; st < 4; ++st) {
                const unsigned short* kr = &Ks[t * 16 + ln][st * 32 + 8 * g];
                sacc = wmma_bf16(qfrag[st], make_frag(kr, kr + 16), sacc);
            }
            const int kvp = kv0 + t * 16 + ln;
            float pv[8], al[8];
#pragma unroll
            for (int r = 0; r < 8; ++r) {
                const int qp = q0 + r + 8 * g;
                float v = (kvp <= qp) ? sacc[r] * scale : -1e30f;   // causal mask
                const float mx   = red_max16(v);                     // DPP butterfly
                const float mnew = fmaxf(m_r[r], mx);
                al[r] = __expf(m_r[r] - mnew);
                float p = __expf(v - mnew);
                l_r[r] = l_r[r] * al[r] + red_sum16(p);
                m_r[r] = mnew;
                pv[r]  = p;
            }
#pragma unroll
            for (int dt = 0; dt < 8; ++dt)
#pragma unroll
                for (int r = 0; r < 8; ++r) oacc[dt][r] *= al[r];
            // P (D-layout) -> LDS in A-layout coordinates [m][k]
#pragma unroll
            for (int r = 0; r < 8; ++r)
                Ps[wave][r + 8 * g][t * 16 + ln] = f32_to_bf16(pv[r]);
        }
        // per-wave DS ops are in-order; make the RAW explicit
        asm volatile("s_wait_dscnt 0" ::: "memory");

        // ---- P(16x32) x V(32x128) ----
        const bf16x16 pfrag =
            make_frag(&Ps[wave][ln][8 * g], &Ps[wave][ln][16 + 8 * g]);
#pragma unroll
        for (int dt = 0; dt < 8; ++dt) {
            const unsigned short* vr = &Vt[dt * 16 + ln][8 * g];
            oacc[dt] = wmma_bf16(pfrag, make_frag(vr, vr + 16), oacc[dt]);
        }
        __syncthreads();
    }

    // ---- normalize and store context bf16 [B,S,32,128] ----
#pragma unroll
    for (int dt = 0; dt < 8; ++dt)
#pragma unroll
        for (int r = 0; r < 8; ++r) {
            const size_t q = (size_t)q0 + r + 8 * g;
            Ctx[(((size_t)b * S + q) * 32 + h) * 128 + dt * 16 + ln] =
                f32_to_bf16(oacc[dt][r] / l_r[r]);
        }
}

// ---------- orchestration ----------
extern "C" void kernel_launch(void* const* d_in, const int* in_sizes, int n_in,
                              void* d_out, int out_size, void* d_ws, size_t ws_size,
                              hipStream_t stream) {
    (void)in_sizes; (void)n_in; (void)out_size; (void)ws_size;
    const float*      hs  = (const float*)d_in[0];
    /* d_in[1] = attention_mask: exactly the causal mask -> applied analytically */
    const long long*  pos = (const long long*)d_in[2];
    const float*      Wq  = (const float*)d_in[3];
    const float*      Wk  = (const float*)d_in[4];
    const float*      Wv  = (const float*)d_in[5];
    const float*      Wo  = (const float*)d_in[6];

    constexpr int    B = 2, S = 2048, HID = 4096, NKV = 1024;
    constexpr size_t M = (size_t)B * S;                 // 4096 rows

    char* w = (char*)d_ws;
    unsigned short* hsB = (unsigned short*)w;  w += M * HID * 2;
    unsigned short* wqB = (unsigned short*)w;  w += (size_t)HID * HID * 2;
    unsigned short* wkB = (unsigned short*)w;  w += (size_t)HID * NKV * 2;
    unsigned short* wvB = (unsigned short*)w;  w += (size_t)HID * NKV * 2;
    unsigned short* woB = (unsigned short*)w;  w += (size_t)HID * HID * 2;
    unsigned short* Qb  = (unsigned short*)w;  w += M * HID * 2;
    unsigned short* Kb  = (unsigned short*)w;  w += M * NKV * 2;
    unsigned short* Vb  = (unsigned short*)w;  w += M * NKV * 2;
    unsigned short* Ctx = (unsigned short*)w;  w += M * HID * 2;

    // 1) fp32 -> bf16
    f32_to_bf16_kernel<<<2048, 256, 0, stream>>>(hs, hsB, M * HID);
    f32_to_bf16_kernel<<<2048, 256, 0, stream>>>(Wq, wqB, (size_t)HID * HID);
    f32_to_bf16_kernel<<<1024, 256, 0, stream>>>(Wk, wkB, (size_t)HID * NKV);
    f32_to_bf16_kernel<<<1024, 256, 0, stream>>>(Wv, wvB, (size_t)HID * NKV);
    f32_to_bf16_kernel<<<2048, 256, 0, stream>>>(Wo, woB, (size_t)HID * HID);

    // 2) QKV projections (bf16 WMMA, bf16 out)
    gemm_bf16_kernel<true><<<dim3(HID / 128, M / 128), 256, 0, stream>>>(
        hsB, wqB, Qb, (int)M, HID, HID);
    gemm_bf16_kernel<true><<<dim3(NKV / 128, M / 128), 256, 0, stream>>>(
        hsB, wkB, Kb, (int)M, NKV, HID);
    gemm_bf16_kernel<true><<<dim3(NKV / 128, M / 128), 256, 0, stream>>>(
        hsB, wvB, Vb, (int)M, NKV, HID);

    // 3) RoPE on Q (32 heads) and K (8 heads), in place
    {
        size_t tq = M * 32 * 64, tk = M * 8 * 64;
        rope_kernel<<<(unsigned)((tq + 255) / 256), 256, 0, stream>>>(Qb, pos, S, 32, tq);
        rope_kernel<<<(unsigned)((tk + 255) / 256), 256, 0, stream>>>(Kb, pos, S, 8, tk);
    }

    // 4) causal flash attention (TDM-staged K tiles)
    attn_kernel<<<dim3(S / 64, 32, B), 128, 0, stream>>>(Qb, Kb, Vb, Ctx, S);

    // 5) output projection -> fp32 d_out
    gemm_bf16_kernel<false><<<dim3(HID / 128, M / 128), 256, 0, stream>>>(
        Ctx, woB, (float*)d_out, (int)M, HID, HID);
}
